// SyntaxGNNEncoder_60559038873899
// MI455X (gfx1250) — matrix operand order repaired and checked by reading, not compiled, not run on hardware
//
#include <hip/hip_runtime.h>
#include <hip/hip_bf16.h>

#define N_NODES 65536
#define N_EDGES 262144
#define N_GRAPHS 2048
#define N_HEADS 4
#define HEAD_DIM 192
#define HID 768
#define N_LAYERS 2
#define NEG_SLOPE 0.2f
#define LN_EPS 1e-5f

typedef unsigned short ushort_t;
typedef __attribute__((ext_vector_type(16))) __bf16 v16bf;
typedef __attribute__((ext_vector_type(8)))  __bf16 v8bf;
typedef __attribute__((ext_vector_type(8)))  float  v8f;

__device__ __forceinline__ ushort_t f2bf(float f) {
  unsigned int u = __float_as_uint(f);
  u += 0x7fffu + ((u >> 16) & 1u);          // round-to-nearest-even
  return (ushort_t)(u >> 16);
}
__device__ __forceinline__ int f2ord(float f) {
  int i = __float_as_int(f);
  return i >= 0 ? i : (i ^ 0x7fffffff);
}
__device__ __forceinline__ float ord2f(int o) {
  int i = (o >= 0) ? o : (o ^ 0x7fffffff);
  return __int_as_float(i);
}
__device__ __forceinline__ float lrelu(float a) { return a > 0.f ? a : NEG_SLOPE * a; }

// --- CDNA5 async global->LDS (ASYNCcnt path), via inline asm -----------------
__device__ __forceinline__ void async_ld_b128(unsigned lds_addr, const void* gaddr) {
  asm volatile("global_load_async_to_lds_b128 %0, %1, off"
               :: "v"(lds_addr), "v"(gaddr) : "memory");
}
__device__ __forceinline__ void wait_async_le4() {
  asm volatile("s_wait_asynccnt 0x4" ::: "memory");
}
__device__ __forceinline__ void wait_async_0() {
  asm volatile("s_wait_asynccnt 0x0" ::: "memory");
}
__device__ __forceinline__ unsigned lds_addr32(const void* p) {
  return (unsigned)(unsigned long long)p;   // low 32 bits of generic addr = LDS offset
}

// ---------------------------------------------------------------------------
// f32 -> bf16 elementwise convert
// ---------------------------------------------------------------------------
__global__ __launch_bounds__(256) void k_convert_bf16(const float* __restrict__ src,
                                                      ushort_t* __restrict__ dst, int n) {
  int i = blockIdx.x * 256 + threadIdx.x;
  if (i < n) dst[i] = f2bf(src[i]);
}

// transpose + convert: dst[n][k] = bf16(src[k][n]) for a 768x768 matrix
__global__ __launch_bounds__(256) void k_tpose_bf16(const float* __restrict__ src,
                                                    ushort_t* __restrict__ dst) {
  int i = blockIdx.x * 256 + threadIdx.x;
  if (i >= HID * HID) return;
  int n = i / HID, k = i - n * HID;
  dst[n * HID + k] = f2bf(src[k * HID + n]);
}

// ---------------------------------------------------------------------------
// WMMA bf16 GEMM: C[M,768] = A[M,768] * W  where Bt[n][k] = W[k][n] (bf16)
// Block tile 128(M) x 128(N), 8 waves (2 along M x 4 along N), each wave owns
// 4x2 accumulators (64x32). K step 32, double-buffered async global->LDS.
// ---------------------------------------------------------------------------
__global__ __launch_bounds__(256) void k_wmma_gemm(const ushort_t* __restrict__ A,
                                                   const ushort_t* __restrict__ Bt,
                                                   float* __restrict__ C,
                                                   const float* __restrict__ bias,
                                                   int M) {
  __shared__ alignas(16) ushort_t As[2][128 * 40];  // padded stride 40 -> no conflicts
  __shared__ alignas(16) ushort_t Bs[2][128 * 40];
  const int bm = blockIdx.x * 128;
  const int bn = blockIdx.y * 128;
  const int tid = threadIdx.x;
  const int lane = tid & 31;
  const int wave = tid >> 5;
  const int wm = wave & 1;            // 2 waves along M (64 rows each)
  const int wn = wave >> 1;           // 4 waves along N (32 cols each)
  const int r15 = lane & 15;
  const int kbA = (lane >> 4) * 8;    // A frag: lanes<16 -> K 0-7/16-23, else 8-15/24-31
  const int kbB = (lane >> 4) * 8;    // B frag: lanes<16 -> K 0-7/16-23, else 8-15/24-31

  v8f acc[4][2];
#pragma unroll
  for (int i = 0; i < 4; ++i)
#pragma unroll
    for (int j = 0; j < 2; ++j) {
      v8f z = {0.f, 0.f, 0.f, 0.f, 0.f, 0.f, 0.f, 0.f};
      acc[i][j] = z;
    }

  // 256 threads stage a 128x32 A tile and a 128x32 B tile: 2 b128 chunks each.
  const int srow = tid >> 1;                 // 0..127
  const int sch = (tid & 1) * 16;            // 0 or 16 elements
  const ushort_t* Ag = A + (size_t)(bm + srow) * HID + sch;
  const ushort_t* Bg = Bt + (size_t)(bn + srow) * HID + sch;
  const unsigned aL = lds_addr32(&As[0][srow * 40 + sch]);
  const unsigned bL = lds_addr32(&Bs[0][srow * 40 + sch]);
  const unsigned bufBytes = (unsigned)sizeof(As[0]);

  // prologue: stage K-step 0 into buffer 0
  async_ld_b128(aL, Ag);
  async_ld_b128(aL + 16, Ag + 8);
  async_ld_b128(bL, Bg);
  async_ld_b128(bL + 16, Bg + 8);

  for (int it = 0; it < HID / 32; ++it) {
    const int buf = it & 1;
    if (it + 1 < HID / 32) {                 // stage next K-step into other buffer
      const int k0 = (it + 1) * 32;
      const unsigned la = aL + (buf ^ 1) * bufBytes;
      const unsigned lb = bL + (buf ^ 1) * bufBytes;
      async_ld_b128(la, Ag + k0);
      async_ld_b128(la + 16, Ag + k0 + 8);
      async_ld_b128(lb, Bg + k0);
      async_ld_b128(lb + 16, Bg + k0 + 8);
      wait_async_le4();                      // older 4 (current buf) are complete
    } else {
      wait_async_0();
    }
    __syncthreads();

    const ushort_t* Asb = As[buf];
    const ushort_t* Bsb = Bs[buf];
    v16bf af[4], bfr[2];
#pragma unroll
    for (int mi = 0; mi < 4; ++mi) {
      int row = wm * 64 + mi * 16 + r15;
      v8bf lo = *(const v8bf*)&Asb[row * 40 + kbA];
      v8bf hi = *(const v8bf*)&Asb[row * 40 + 16 + kbA];
      af[mi] = __builtin_shufflevector(lo, hi, 0, 1, 2, 3, 4, 5, 6, 7, 8, 9, 10, 11, 12, 13, 14, 15);
    }
#pragma unroll
    for (int ni = 0; ni < 2; ++ni) {
      int col = wn * 32 + ni * 16 + r15;
      v8bf b0f = *(const v8bf*)&Bsb[col * 40 + kbB];
      v8bf b1f = *(const v8bf*)&Bsb[col * 40 + 16 + kbB];
      bfr[ni] = __builtin_shufflevector(b0f, b1f, 0, 1, 2, 3, 4, 5, 6, 7, 8, 9, 10, 11, 12, 13, 14, 15);
    }
#pragma unroll
    for (int mi = 0; mi < 4; ++mi)
#pragma unroll
      for (int ni = 0; ni < 2; ++ni)
        acc[mi][ni] = __builtin_amdgcn_wmma_f32_16x16x32_bf16(
            false, af[mi], false, bfr[ni], (short)0, acc[mi][ni], false, false);
    __syncthreads();                          // all waves done with buf before overwrite
  }

  const int mbase = (lane >> 4) * 8;
#pragma unroll
  for (int mi = 0; mi < 4; ++mi)
#pragma unroll
    for (int ni = 0; ni < 2; ++ni) {
      int col = bn + wn * 32 + ni * 16 + r15;
      float bv = bias ? bias[col] : 0.f;
#pragma unroll
      for (int i = 0; i < 8; ++i) {
        int row = bm + wm * 64 + mi * 16 + mbase + i;
        C[(size_t)row * HID + col] = acc[mi][ni][i] + bv;
      }
    }
}

// ---------------------------------------------------------------------------
// Per (node, head): alpha_src/alpha_dst dot products + self-loop max init.
// One wave per (n,h); 192 dims -> 6 elements per lane.
// ---------------------------------------------------------------------------
__global__ __launch_bounds__(256) void k_alpha(const float* __restrict__ xw,
                                               const float* __restrict__ att_s,
                                               const float* __restrict__ att_d,
                                               float* __restrict__ asrc,
                                               float* __restrict__ adst,
                                               int* __restrict__ mord) {
  int gw = (blockIdx.x * 256 + threadIdx.x) >> 5;
  int lane = threadIdx.x & 31;
  if (gw >= N_NODES * N_HEADS) return;
  int n = gw >> 2, h = gw & 3;
  float ss = 0.f, sd = 0.f;
#pragma unroll
  for (int t = 0; t < 6; ++t) {
    int d = t * 32 + lane;
    float x = xw[(size_t)n * HID + h * HEAD_DIM + d];
    ss += x * att_s[h * HEAD_DIM + d];
    sd += x * att_d[h * HEAD_DIM + d];
  }
#pragma unroll
  for (int off = 16; off > 0; off >>= 1) {
    ss += __shfl_down(ss, off, 32);
    sd += __shfl_down(sd, off, 32);
  }
  if (lane == 0) {
    asrc[n * 4 + h] = ss;
    adst[n * 4 + h] = sd;
    mord[n * 4 + h] = f2ord(lrelu(ss + sd));   // self-loop seeds the segment max
  }
}

// edge pass 1: segment max of leaky(e) per destination (ordered-int atomicMax)
__global__ __launch_bounds__(256) void k_edge_max(const int* __restrict__ src,
                                                  const int* __restrict__ dst,
                                                  const float* __restrict__ asrc,
                                                  const float* __restrict__ adst,
                                                  int* __restrict__ mord) {
  int e = blockIdx.x * 256 + threadIdx.x;
  if (e >= N_EDGES) return;
  int j = src[e], i = dst[e];
#pragma unroll
  for (int h = 0; h < 4; ++h) {
    float a = lrelu(asrc[j * 4 + h] + adst[i * 4 + h]);
    atomicMax(&mord[i * 4 + h], f2ord(a));
  }
}

// self-loop: denom[n,h] = exp(e_self - m)
__global__ __launch_bounds__(256) void k_self_denom(const float* __restrict__ asrc,
                                                    const float* __restrict__ adst,
                                                    const int* __restrict__ mord,
                                                    float* __restrict__ denom) {
  int i = blockIdx.x * 256 + threadIdx.x;
  if (i >= N_NODES * 4) return;
  float e = lrelu(asrc[i] + adst[i]);
  denom[i] = __expf(e - ord2f(mord[i]));
}

// self-loop contribution initializes the aggregation buffer (no zero-init needed)
__global__ __launch_bounds__(256) void k_self_out(const float* __restrict__ xw,
                                                  const float* __restrict__ denom,
                                                  float* __restrict__ aggr) {
  int i = blockIdx.x * 256 + threadIdx.x;
  if (i >= N_NODES * HID) return;
  int n = i / HID, c = i - n * HID;
  aggr[i] = denom[n * 4 + c / HEAD_DIM] * xw[i];
}

// edge pass 2: one wave per edge; atomicAdd exp-weighted features into aggr
__global__ __launch_bounds__(256) void k_edge_aggr(const int* __restrict__ src,
                                                   const int* __restrict__ dst,
                                                   const float* __restrict__ asrc,
                                                   const float* __restrict__ adst,
                                                   const int* __restrict__ mord,
                                                   const float* __restrict__ xw,
                                                   float* __restrict__ denom,
                                                   float* __restrict__ aggr) {
  int gw = (blockIdx.x * 256 + threadIdx.x) >> 5;
  int lane = threadIdx.x & 31;
  if (gw >= N_EDGES) return;
  int j = src[gw], i = dst[gw];
  float ex[4];
#pragma unroll
  for (int h = 0; h < 4; ++h) {
    float a = lrelu(asrc[j * 4 + h] + adst[i * 4 + h]);
    ex[h] = __expf(a - ord2f(mord[i * 4 + h]));
  }
  if (lane < 4) {
    float v = (lane == 0) ? ex[0] : (lane == 1) ? ex[1] : (lane == 2) ? ex[2] : ex[3];
    atomicAdd(&denom[i * 4 + lane], v);
  }
#pragma unroll
  for (int t = 0; t < 24; ++t) {          // 24*32 = 768; head = t/6 is unroll-constant
    int c = t * 32 + lane;
    atomicAdd(&aggr[(size_t)i * HID + c], ex[t / 6] * xw[(size_t)j * HID + c]);
  }
}

// divide by denom + bias + LayerNorm + ReLU + residual (block per node)
__global__ __launch_bounds__(256) void k_ln_relu_res(const float* __restrict__ aggr,
                                                     const float* __restrict__ denom,
                                                     const float* __restrict__ bias,
                                                     const float* __restrict__ gamma,
                                                     const float* __restrict__ beta,
                                                     float* __restrict__ h) {
  __shared__ float ssum[8], ssq[8];
  int n = blockIdx.x, t = threadIdx.x;
  float v[3], s = 0.f, q = 0.f;
#pragma unroll
  for (int i = 0; i < 3; ++i) {
    int c = t + i * 256;
    float x = aggr[(size_t)n * HID + c] / denom[n * 4 + c / HEAD_DIM] + bias[c];
    v[i] = x; s += x; q += x * x;
  }
#pragma unroll
  for (int off = 16; off > 0; off >>= 1) {
    s += __shfl_down(s, off, 32);
    q += __shfl_down(q, off, 32);
  }
  int wid = t >> 5, lane = t & 31;
  if (lane == 0) { ssum[wid] = s; ssq[wid] = q; }
  __syncthreads();
  if (t == 0) {
    float a = 0.f, b = 0.f;
#pragma unroll
    for (int i = 0; i < 8; ++i) { a += ssum[i]; b += ssq[i]; }
    ssum[0] = a; ssq[0] = b;
  }
  __syncthreads();
  float mean = ssum[0] * (1.f / HID);
  float var = ssq[0] * (1.f / HID) - mean * mean;
  float rs = rsqrtf(var + LN_EPS);
#pragma unroll
  for (int i = 0; i < 3; ++i) {
    int c = t + i * 256;
    float y = (v[i] - mean) * rs * gamma[c] + beta[c];
    y = fmaxf(y, 0.f);
    h[(size_t)n * HID + c] = y + h[(size_t)n * HID + c];
  }
}

// ---------------------------------------------------------------------------
// Global mean pool + readout prep
// ---------------------------------------------------------------------------
__global__ __launch_bounds__(256) void k_pool_init(float* __restrict__ hg,
                                                   float* __restrict__ cnt) {
  int i = blockIdx.x * 256 + threadIdx.x;
  if (i < N_GRAPHS * HID) hg[i] = 0.f;
  if (i < N_GRAPHS) cnt[i] = 0.f;
}

__global__ __launch_bounds__(256) void k_pool(const float* __restrict__ h,
                                              const int* __restrict__ batch_vec,
                                              float* __restrict__ hg,
                                              float* __restrict__ cnt) {
  int i = blockIdx.x * 256 + threadIdx.x;
  if (i >= N_NODES * HID) return;
  int n = i / HID, c = i - n * HID;
  int g = batch_vec[n];
  atomicAdd(&hg[(size_t)g * HID + c], h[i]);
  if (c == 0) atomicAdd(&cnt[g], 1.0f);
}

__global__ __launch_bounds__(256) void k_pool_norm(const float* __restrict__ hg,
                                                   const float* __restrict__ cnt,
                                                   ushort_t* __restrict__ hgb) {
  int i = blockIdx.x * 256 + threadIdx.x;
  if (i >= N_GRAPHS * HID) return;
  int g = i / HID;
  hgb[i] = f2bf(hg[i] / fmaxf(cnt[g], 1.0f));
}

// ---------------------------------------------------------------------------
extern "C" void kernel_launch(void* const* d_in, const int* in_sizes, int n_in,
                              void* d_out, int out_size, void* d_ws, size_t ws_size,
                              hipStream_t stream) {
  const float* x = (const float*)d_in[0];
  const int* ei = (const int*)d_in[1];
  const int* src = ei;                  // edge_index[0]
  const int* dst = ei + N_EDGES;        // edge_index[1]
  const int* batch_vec = (const int*)d_in[2];
  const float* W = (const float*)d_in[3];
  const float* att_src = (const float*)d_in[4];
  const float* att_dst = (const float*)d_in[5];
  const float* bias = (const float*)d_in[6];
  const float* ln_gamma = (const float*)d_in[7];
  const float* ln_beta = (const float*)d_in[8];
  const float* out_w = (const float*)d_in[9];
  const float* out_b = (const float*)d_in[10];
  float* out = (float*)d_out;

  char* ws = (char*)d_ws;
  size_t off = 0;
  const size_t NF = (size_t)N_NODES * HID;
  float* h = (float*)(ws + off);      off += NF * 4;                     // 192 MB
  float* xw = (float*)(ws + off);     off += NF * 4;                     // 192 MB
  float* aggr = (float*)(ws + off);   off += NF * 4;                     // 192 MB
  ushort_t* hb = (ushort_t*)(ws + off); off += NF * 2;                   //  96 MB
  float* asrc = (float*)(ws + off);   off += (size_t)N_NODES * 4 * 4;
  float* adst = (float*)(ws + off);   off += (size_t)N_NODES * 4 * 4;
  int* mord = (int*)(ws + off);       off += (size_t)N_NODES * 4 * 4;
  float* denom = (float*)(ws + off);  off += (size_t)N_NODES * 4 * 4;
  ushort_t* Wt = (ushort_t*)(ws + off);  off += (size_t)N_LAYERS * HID * HID * 2;
  ushort_t* OWt = (ushort_t*)(ws + off); off += (size_t)HID * HID * 2;
  float* hg = (float*)(ws + off);     off += (size_t)N_GRAPHS * HID * 4;
  float* cnt = (float*)(ws + off);    off += (size_t)N_GRAPHS * 4;
  ushort_t* hgb = (ushort_t*)(ws + off); off += (size_t)N_GRAPHS * HID * 2;

  // h = x
  hipMemcpyAsync(h, x, NF * 4, hipMemcpyDeviceToDevice, stream);

  // transpose+convert weights to bf16 (Bt[n][k] layout)
  const int TP = HID * HID / 256;
  k_tpose_bf16<<<TP, 256, 0, stream>>>(W, Wt);
  k_tpose_bf16<<<TP, 256, 0, stream>>>(W + (size_t)HID * HID, Wt + (size_t)HID * HID);
  k_tpose_bf16<<<TP, 256, 0, stream>>>(out_w, OWt);

  const int NFB = (int)(NF / 256);                 // 196608 blocks
  const int NH4 = N_NODES * 4 / 256;               // 1024
  const int NHW = N_NODES * N_HEADS * 32 / 256;    // 32768 (wave per (n,h))
  const int EB = N_EDGES / 256;                    // 1024
  const int EW = N_EDGES * 32 / 256;               // 32768 (wave per edge)

  for (int l = 0; l < N_LAYERS; ++l) {
    k_convert_bf16<<<NFB, 256, 0, stream>>>(h, hb, (int)NF);
    k_wmma_gemm<<<dim3(N_NODES / 128, HID / 128), 256, 0, stream>>>(
        hb, Wt + (size_t)l * HID * HID, xw, nullptr, N_NODES);
    k_alpha<<<NHW, 256, 0, stream>>>(xw, att_src + (size_t)l * HID,
                                     att_dst + (size_t)l * HID, asrc, adst, mord);
    k_edge_max<<<EB, 256, 0, stream>>>(src, dst, asrc, adst, mord);
    k_self_denom<<<NH4, 256, 0, stream>>>(asrc, adst, mord, denom);
    k_self_out<<<NFB, 256, 0, stream>>>(xw, denom, aggr);
    k_edge_aggr<<<EW, 256, 0, stream>>>(src, dst, asrc, adst, mord, xw, denom, aggr);
    k_ln_relu_res<<<N_NODES, 256, 0, stream>>>(aggr, denom, bias + (size_t)l * HID,
                                               ln_gamma + (size_t)l * HID,
                                               ln_beta + (size_t)l * HID, h);
  }

  const int GB = N_GRAPHS * HID / 256;             // 6144
  k_pool_init<<<GB, 256, 0, stream>>>(hg, cnt);
  k_pool<<<NFB, 256, 0, stream>>>(h, batch_vec, hg, cnt);
  k_pool_norm<<<GB, 256, 0, stream>>>(hg, cnt, hgb);
  k_wmma_gemm<<<dim3(N_GRAPHS / 128, HID / 128), 256, 0, stream>>>(
      hgb, OWt, out, out_b, N_GRAPHS);
}
